// DifferentialMultiHeadAttention_24300924961408
// MI455X (gfx1250) — matrix operand order repaired and checked by reading, not compiled, not run on hardware
//
#include <hip/hip_runtime.h>

// ---------------------------------------------------------------------------
// DifferentialMultiHeadAttention for MI455X (gfx1250, wave32, WMMA)
// B=4, S=1024, D=1024, H=16 (8 card + 4 deck + 4 causal), DH=64
// ---------------------------------------------------------------------------

typedef __attribute__((ext_vector_type(16))) _Float16 v16h;
typedef __attribute__((ext_vector_type(8)))  _Float16 v8h;
typedef __attribute__((ext_vector_type(4)))  _Float16 v4h;
typedef __attribute__((ext_vector_type(8)))  float    v8f;

#define Bc 4
#define Sc 1024
#define Dc 1024
#define Hc 16
#define DHc 64

union AFrag { v16h v; v8h h[2]; };

// A-fragment (16x32 f16, contraction K=32) from row-major [M][K] source.
// Lane L<16 : row = row0+L,  halfs 0-15 = K 0..15 (contiguous)
// Lane L>=16: row = row0+L-16, halfs 0-15 = K 16..31 (contiguous)
__device__ inline v16h load_a16x32(const _Float16* __restrict__ base, int ld,
                                   int row0, int k0, int lane) {
  int r = lane & 15, hf = lane >> 4;
  const _Float16* p = base + (size_t)(row0 + r) * ld + k0 + hf * 16;
  AFrag a;
  a.h[0] = *(const v8h*)(p);
  a.h[1] = *(const v8h*)(p + 8);
  return a.v;
}

// B-fragment (32x16 f16) from a [N][K]-major source (i.e. B^T row-major).
// Lane L<16 : col N = n0+L,  halfs = K 0..15 (contiguous)
// Lane L>=16: col N = n0+L-16, halfs = K 16..31 (contiguous)
__device__ inline v16h load_b32x16(const _Float16* __restrict__ bt, int ld,
                                   int n0, int k0, int lane) {
  int r = lane & 15, hf = lane >> 4;
  return *(const v16h*)(bt + (size_t)(n0 + r) * ld + k0 + hf * 16);
}

__device__ inline v8f wmma_f16(v16h a, v16h b, v8f c) {
  return __builtin_amdgcn_wmma_f32_16x16x32_f16(false, a, false, b,
                                                (short)0, c, false, false);
}

// ---------------------------------------------------------------------------
// fp32 -> f16 convert (4 elements / thread)
// ---------------------------------------------------------------------------
__global__ __launch_bounds__(256) void cvt_f32_f16(const float* __restrict__ in,
                                                   _Float16* __restrict__ out,
                                                   int n) {
  int i = (blockIdx.x * 256 + threadIdx.x) * 4;
  if (i >= n) return;
  float4 f = *(const float4*)(in + i);
  v4h h = { (_Float16)f.x, (_Float16)f.y, (_Float16)f.z, (_Float16)f.w };
  *(v4h*)(out + i) = h;
}

// ---------------------------------------------------------------------------
// QKV projection: qkv(4096,3072) = xh(4096,1024) @ Wqkv^T, + bias.
// Each wave computes a 64x64 tile (4x4 WMMA accumulators, B streamed).
// Epilogue scatters into Q,K (B,H,S,DH) f16 and Vt (B,H,DH,S) f16.
// ---------------------------------------------------------------------------
__global__ __launch_bounds__(256) void qkv_gemm(
    const _Float16* __restrict__ xh, const _Float16* __restrict__ w,
    const float* __restrict__ bias,
    _Float16* __restrict__ Q, _Float16* __restrict__ K,
    _Float16* __restrict__ Vt) {
  const int NT = (3 * Dc) / 64;  // 48 tiles along N
  int wave = (blockIdx.x * blockDim.x + threadIdx.x) >> 5;
  int lane = threadIdx.x & 31;
  int mt = wave / NT, nt = wave % NT;
  int m0 = mt * 64, n0 = nt * 64;

  v8f c[4][4] = {};
  for (int k0 = 0; k0 < Dc; k0 += 32) {
    // prefetch next k-chunk (global_prefetch_b8); 192MB L2 holds everything,
    // this pulls the next tile toward the WGP while WMMAs run
    if (k0 + 32 < Dc) {
      __builtin_prefetch(xh + (size_t)(m0 + (lane & 63)) * Dc + k0 + 32, 0, 3);
      __builtin_prefetch(w  + (size_t)(n0 + (lane & 63)) * Dc + k0 + 32, 0, 3);
    }
    v16h a[4];
#pragma unroll
    for (int i = 0; i < 4; ++i)
      a[i] = load_a16x32(xh, Dc, m0 + 16 * i, k0, lane);
#pragma unroll
    for (int j = 0; j < 4; ++j) {
      v16h bf = load_b32x16(w, Dc, n0 + 16 * j, k0, lane);
#pragma unroll
      for (int i = 0; i < 4; ++i) c[i][j] = wmma_f16(a[i], bf, c[i][j]);
    }
  }

  int rr = lane & 15, hf = lane >> 4;
#pragma unroll
  for (int i = 0; i < 4; ++i)
#pragma unroll
    for (int j = 0; j < 4; ++j)
#pragma unroll
      for (int r = 0; r < 8; ++r) {
        int row = m0 + i * 16 + r + hf * 8;   // token index in [0, B*S)
        int col = n0 + j * 16 + rr;           // column in [0, 3D)
        float v = c[i][j][r] + bias[col];
        _Float16 hv = (_Float16)v;
        int which = col >> 10;                // 0=q, 1=k, 2=v
        int d = col & (Dc - 1);
        int h = d >> 6, dh = d & 63;
        int b = row >> 10, s = row & (Sc - 1);
        size_t bh = (size_t)(b * Hc + h);
        if (which == 0)      Q[(bh * Sc + s) * DHc + dh] = hv;
        else if (which == 1) K[(bh * Sc + s) * DHc + dh] = hv;
        else                 Vt[(bh * DHc + dh) * Sc + s] = hv;
      }
}

// ---------------------------------------------------------------------------
// Flash attention: one wave = 16 queries of one (b,h). Keys in chunks of 32.
// Card heads (h<8): mask + time-decay bias. Deck (8..11): mask. Global: causal.
// ---------------------------------------------------------------------------
__global__ __launch_bounds__(256) void attn_kernel(
    const _Float16* __restrict__ Q, const _Float16* __restrict__ Kk,
    const _Float16* __restrict__ Vt,
    const unsigned char* __restrict__ card_mask,
    const unsigned char* __restrict__ deck_mask,
    const float* __restrict__ time_diff,
    const float* __restrict__ td_weight,
    const float* __restrict__ td_decay_raw,
    _Float16* __restrict__ attnout) {
  __shared__ __align__(64) _Float16 pbuf[8][16][32];  // per-wave P staging

  int wv = threadIdx.x >> 5;
  int lane = threadIdx.x & 31;
  int gwave = blockIdx.x * 8 + wv;
  int mt = gwave & 63;           // S/16 tiles
  int bh = gwave >> 6;           // b*H + h
  int b = bh >> 4, h = bh & 15;
  int q0 = mt * 16;

  const _Float16* Qbh = Q  + (size_t)bh * Sc * DHc;
  const _Float16* Kbh = Kk + (size_t)bh * Sc * DHc;
  const _Float16* Vbh = Vt + (size_t)bh * DHc * Sc;

  int type = (h < 8) ? 0 : (h < 12 ? 1 : 2);
  float tw = 0.f, dec = 0.f;
  if (type == 0) {
    tw = td_weight[h];
    float xr = td_decay_raw[h];
    dec = (xr > 20.f) ? xr : log1pf(__expf(xr));   // softplus
  }
  const unsigned char* cm =
      (type == 0) ? card_mask + (size_t)b * Sc * Sc
                  : deck_mask + (size_t)b * Sc * Sc;
  const float* tdp = time_diff + (size_t)b * Sc * Sc;

  // Q fragments for the whole key loop (contraction DH=64 -> 2 A-frags)
  v16h qa0 = load_a16x32(Qbh, DHc, q0, 0,  lane);
  v16h qa1 = load_a16x32(Qbh, DHc, q0, 32, lane);

  v8f o0 = {}, o1 = {}, o2 = {}, o3 = {};
  float rm[8], rl[8];
#pragma unroll
  for (int r = 0; r < 8; ++r) { rm[r] = -1e30f; rl[r] = 0.f; }

  int rr = lane & 15, hf = lane >> 4;

  for (int k0 = 0; k0 < Sc; k0 += 32) {
    // ---- scores: 16x32 tile = two 16x16 C-frags, 2 WMMAs each ----
    v16h kb00 = load_b32x16(Kbh, DHc, k0,      0,  lane);
    v16h kb01 = load_b32x16(Kbh, DHc, k0,      32, lane);
    v16h kb10 = load_b32x16(Kbh, DHc, k0 + 16, 0,  lane);
    v16h kb11 = load_b32x16(Kbh, DHc, k0 + 16, 32, lane);
    v8f s0 = {}, s1 = {};
    s0 = wmma_f16(qa0, kb00, s0);
    s0 = wmma_f16(qa1, kb01, s0);
    s1 = wmma_f16(qa0, kb10, s1);
    s1 = wmma_f16(qa1, kb11, s1);

    // ---- scale + mask + bias (fp32, C-fragment layout) ----
#pragma unroll
    for (int r = 0; r < 8; ++r) {
      int qrow = q0 + r + hf * 8;
      int c0 = k0 + rr, c1 = k0 + 16 + rr;
      float v0 = s0[r] * 0.125f, v1 = s1[r] * 0.125f;
      if (type == 2) {
        if (c0 > qrow) v0 = -1e30f;
        if (c1 > qrow) v1 = -1e30f;
      } else {
        size_t base = (size_t)qrow * Sc;
        if (!cm[base + c0]) v0 = -1e30f;
        if (!cm[base + c1]) v1 = -1e30f;
        if (type == 0) {
          v0 += tw * __expf(-dec * tdp[base + c0]);
          v1 += tw * __expf(-dec * tdp[base + c1]);
        }
      }
      s0[r] = v0; s1[r] = v1;
    }

    // ---- online softmax (per-row stats live per VGPR slot + lane-half) ----
#pragma unroll
    for (int r = 0; r < 8; ++r) {
      float mx = fmaxf(s0[r], s1[r]);
#pragma unroll
      for (int off = 1; off < 16; off <<= 1)
        mx = fmaxf(mx, __shfl_xor(mx, off, 32));
      float mnew = fmaxf(rm[r], mx);
      float scale = __expf(rm[r] - mnew);
      float e0 = __expf(s0[r] - mnew);
      float e1 = __expf(s1[r] - mnew);
      float sum = e0 + e1;
#pragma unroll
      for (int off = 1; off < 16; off <<= 1)
        sum += __shfl_xor(sum, off, 32);
      rl[r] = rl[r] * scale + sum;
      rm[r] = mnew;
      o0[r] *= scale; o1[r] *= scale; o2[r] *= scale; o3[r] *= scale;
      s0[r] = e0; s1[r] = e1;
    }

    // ---- P: C-layout -> LDS [16][32] -> A-fragment ----
#pragma unroll
    for (int r = 0; r < 8; ++r) {
      pbuf[wv][r + hf * 8][rr]      = (_Float16)s0[r];
      pbuf[wv][r + hf * 8][16 + rr] = (_Float16)s1[r];
    }
    v16h pa = load_a16x32(&pbuf[wv][0][0], 32, 0, 0, lane);

    // ---- O += P @ V : Vt is [DH][S] so B-frags are contiguous ----
    v16h vb0 = load_b32x16(Vbh, Sc, 0,  k0, lane);
    v16h vb1 = load_b32x16(Vbh, Sc, 16, k0, lane);
    v16h vb2 = load_b32x16(Vbh, Sc, 32, k0, lane);
    v16h vb3 = load_b32x16(Vbh, Sc, 48, k0, lane);
    o0 = wmma_f16(pa, vb0, o0);
    o1 = wmma_f16(pa, vb1, o1);
    o2 = wmma_f16(pa, vb2, o2);
    o3 = wmma_f16(pa, vb3, o3);
  }

  // ---- normalize + store attn output as (B,S,D) f16 ----
#pragma unroll
  for (int r = 0; r < 8; ++r) {
    int qrow = q0 + r + hf * 8;
    size_t base = ((size_t)(b * Sc + qrow)) * Dc + h * DHc;
    float inv = 1.0f / rl[r];
    attnout[base +  0 + rr] = (_Float16)(o0[r] * inv);
    attnout[base + 16 + rr] = (_Float16)(o1[r] * inv);
    attnout[base + 32 + rr] = (_Float16)(o2[r] * inv);
    attnout[base + 48 + rr] = (_Float16)(o3[r] * inv);
  }
}

// ---------------------------------------------------------------------------
// Output projection: out(4096,1024) = attnout @ Wout^T + bias (fp32 out)
// 64x64 per-wave tile, B streamed.
// ---------------------------------------------------------------------------
__global__ __launch_bounds__(256) void out_gemm(
    const _Float16* __restrict__ ah, const _Float16* __restrict__ w,
    const float* __restrict__ bias, float* __restrict__ out) {
  const int NT = Dc / 64;  // 16 tiles along N
  int wave = (blockIdx.x * blockDim.x + threadIdx.x) >> 5;
  int lane = threadIdx.x & 31;
  int mt = wave / NT, nt = wave % NT;
  int m0 = mt * 64, n0 = nt * 64;

  v8f c[4][4] = {};
  for (int k0 = 0; k0 < Dc; k0 += 32) {
    if (k0 + 32 < Dc) {
      __builtin_prefetch(ah + (size_t)(m0 + (lane & 63)) * Dc + k0 + 32, 0, 3);
      __builtin_prefetch(w  + (size_t)(n0 + (lane & 63)) * Dc + k0 + 32, 0, 3);
    }
    v16h a[4];
#pragma unroll
    for (int i = 0; i < 4; ++i)
      a[i] = load_a16x32(ah, Dc, m0 + 16 * i, k0, lane);
#pragma unroll
    for (int j = 0; j < 4; ++j) {
      v16h bf = load_b32x16(w, Dc, n0 + 16 * j, k0, lane);
#pragma unroll
      for (int i = 0; i < 4; ++i) c[i][j] = wmma_f16(a[i], bf, c[i][j]);
    }
  }

  int rr = lane & 15, hf = lane >> 4;
#pragma unroll
  for (int i = 0; i < 4; ++i)
#pragma unroll
    for (int j = 0; j < 4; ++j)
#pragma unroll
      for (int r = 0; r < 8; ++r) {
        int row = m0 + i * 16 + r + hf * 8;
        int col = n0 + j * 16 + rr;
        out[(size_t)row * Dc + col] = c[i][j][r] + bias[col];
      }
}

// ---------------------------------------------------------------------------
extern "C" void kernel_launch(void* const* d_in, const int* in_sizes, int n_in,
                              void* d_out, int out_size, void* d_ws,
                              size_t ws_size, hipStream_t stream) {
  const float* x            = (const float*)d_in[0];
  // d_in[1] = causal_mask (computed arithmetically, unused)
  const unsigned char* card = (const unsigned char*)d_in[2];
  const unsigned char* deck = (const unsigned char*)d_in[3];
  const float* tdiff        = (const float*)d_in[4];
  const float* wqkv         = (const float*)d_in[5];
  const float* bqkv         = (const float*)d_in[6];
  const float* wout         = (const float*)d_in[7];
  const float* bout         = (const float*)d_in[8];
  const float* tdw          = (const float*)d_in[9];
  const float* tdd          = (const float*)d_in[10];
  float* out                = (float*)d_out;

  char* ws = (char*)d_ws;
  const int NTOK = Bc * Sc;           // 4096
  _Float16* xh    = (_Float16*)ws;                 ws += (size_t)NTOK * Dc * 2;
  _Float16* wqkvh = (_Float16*)ws;                 ws += (size_t)3 * Dc * Dc * 2;
  _Float16* wouth = (_Float16*)ws;                 ws += (size_t)Dc * Dc * 2;
  _Float16* Q     = (_Float16*)ws;                 ws += (size_t)NTOK * Dc * 2;
  _Float16* K     = (_Float16*)ws;                 ws += (size_t)NTOK * Dc * 2;
  _Float16* Vt    = (_Float16*)ws;                 ws += (size_t)NTOK * Dc * 2;
  _Float16* ao    = (_Float16*)ws;

  int nx = NTOK * Dc;                 // 4M
  int nwq = 3 * Dc * Dc;              // 3M
  int nwo = Dc * Dc;                  // 1M
  cvt_f32_f16<<<nx / 1024, 256, 0, stream>>>(x, xh, nx);
  cvt_f32_f16<<<nwq / 1024, 256, 0, stream>>>(wqkv, wqkvh, nwq);
  cvt_f32_f16<<<nwo / 1024, 256, 0, stream>>>(wout, wouth, nwo);

  // QKV GEMM: (4096/64) * (3072/64) = 64*48 = 3072 waves -> 384 blocks
  qkv_gemm<<<384, 256, 0, stream>>>(xh, wqkvh, bqkv, Q, K, Vt);

  // Attention: B*H*(S/16) = 4096 waves -> 512 blocks
  attn_kernel<<<512, 256, 0, stream>>>(Q, K, Vt, card, deck, tdiff, tdw, tdd, ao);

  // Out projection: (4096/64)*(1024/64) = 64*16 = 1024 waves -> 128 blocks
  out_gemm<<<128, 256, 0, stream>>>(ao, wouth, bout, out);
}
